// MultiHeadAttention_86500641342053
// MI455X (gfx1250) — compile-verified, matrix operand order checked by reading, not run on hardware
//
#include <hip/hip_runtime.h>
#include <hip/hip_bf16.h>
#include <math.h>

// Problem constants (from reference): B=4, S=2048, D=1024, H=16, HD=64
static constexpr int cB = 4;
static constexpr int cS = 2048;
static constexpr int cD = 1024;
static constexpr int cH = 16;
static constexpr int cHD = 64;

// ---------------- WMMA helpers (CDNA5 wave32, 16x16x32 bf16) ----------------

typedef __attribute__((ext_vector_type(16))) __bf16 v16bf;
typedef __attribute__((ext_vector_type(8)))  float  v8f;
typedef __attribute__((ext_vector_type(4)))  unsigned u32x4;
typedef __attribute__((ext_vector_type(8)))  unsigned u32x8;

union FragU { v16bf v; unsigned u[8]; };

// float -> bf16 bits, round-to-nearest-even
__device__ __forceinline__ unsigned short f2bf(float x) {
  unsigned u = __builtin_bit_cast(unsigned, x);
  unsigned r = u + 0x7FFFu + ((u >> 16) & 1u);
  return (unsigned short)(r >> 16);
}

// Fragment gather for a 16x32 bf16 operand from a K-contiguous LDS tile
// (row-major [row][k], stride ld ushorts). ISA 7.12.2 16-bit A layout:
//   lanes 0-15 : row = lane,    K pairs {0..7, 16..23}
//   lanes 16-31: row = lane-16, K pairs {8..15, 24..31}
// Each half is 4 consecutive dwords -> compiler merges to ds_load_b128.
__device__ __forceinline__ v16bf frag_row(const unsigned short* p, int ld, int lane) {
  int m  = lane & 15;
  int kb = (lane >> 4) << 3;
  FragU f;
#pragma unroll
  for (int i = 0; i < 8; ++i) {
    int k0 = (i < 4) ? (kb + 2 * i) : (16 + kb + 2 * (i - 4));
    f.u[i] = *(const unsigned*)(p + m * ld + k0);
  }
  return f.v;
}

__device__ __forceinline__ v8f wmma_bf16(v16bf a, v16bf b, v8f c) {
  return __builtin_amdgcn_wmma_f32_16x16x32_bf16(
      /*neg_a=*/false, a, /*neg_b=*/false, b,
      /*c_mod=*/(short)0, c, /*reuse_a=*/false, /*reuse_b=*/false);
}

// ---------------- Tensor Data Mover: 2D bf16 tile Global -> LDS ----------------
// Builds a D# (ISA 8.3/8.4) for a rows x tileW tile of 2-byte elements with row
// stride `strideElems`, destination LDS offset `ldsOff`, and issues
// TENSOR_LOAD_TO_LDS with the 2-group (<=2D) form. Caller must be a single wave
// and drain with s_wait_tensorcnt before the workgroup barrier.
__device__ __forceinline__ void tdm_load_2d(const void* gaddr, unsigned ldsOff,
                                            unsigned rows, unsigned tileW,
                                            unsigned strideElems,
                                            unsigned tensorD0, unsigned tensorD1) {
  unsigned long long ga = (unsigned long long)(uintptr_t)gaddr;
  u32x4 g0;
  g0[0] = 1u;                                    // count=1, user descriptor
  g0[1] = ldsOff;                                // lds_addr (bytes)
  g0[2] = (unsigned)ga;                          // global_addr[31:0]
  g0[3] = (unsigned)((ga >> 32) & 0x01FFFFFFu)   // global_addr[56:32]
          | (2u << 30);                          // type=2 ("image")
  u32x8 g1;
  g1[0] = 1u << 16;                              // data_size=1 -> 2 bytes
  g1[1] = (tensorD0 & 0xFFFFu) << 16;            // tensor_dim0[15:0]
  g1[2] = (tensorD0 >> 16) | ((tensorD1 & 0xFFFFu) << 16);  // dim0 hi | dim1 lo
  g1[3] = (tensorD1 >> 16) | (tileW << 16);      // dim1 hi | tile_dim0
  g1[4] = rows;                                  // tile_dim1 (tile_dim2=0)
  g1[5] = strideElems;                           // tensor_dim0_stride[31:0]
  g1[6] = 0;                                     // dim0_stride hi | dim1_stride lo
  g1[7] = 0;
  asm volatile("tensor_load_to_lds %0, %1" :: "s"(g0), "s"(g1) : "memory");
}

__device__ __forceinline__ unsigned lds_off(const void* p) {
  return (unsigned)(uintptr_t)p;  // LDS aperture: addr[31:0] is the LDS byte address
}

// ---------------- elementwise f32 -> bf16 convert ----------------

__global__ __launch_bounds__(256) void cvt_f32_bf16_kernel(const float* __restrict__ in,
                                                           unsigned short* __restrict__ out,
                                                           int n) {
  int i = blockIdx.x * 256 + threadIdx.x;
  if (i < n) out[i] = f2bf(in[i]);
}

// ---------------- f32 [R,C] -> bf16 transposed [C,R] (LDS-tiled) ----------------

__global__ __launch_bounds__(256) void cvt_t_kernel(const float* __restrict__ in,
                                                    unsigned short* __restrict__ out,
                                                    int R, int C) {
  __shared__ unsigned short tile[32][33];
  const int c0 = blockIdx.x * 32, r0 = blockIdx.y * 32;
  const int tx = threadIdx.x & 31, ty = threadIdx.x >> 5;  // 32 x 8
#pragma unroll
  for (int i = 0; i < 32; i += 8)
    tile[ty + i][tx] = f2bf(in[(size_t)(r0 + ty + i) * C + c0 + tx]);
  __syncthreads();
#pragma unroll
  for (int i = 0; i < 32; i += 8)
    out[(size_t)(c0 + ty + i) * R + r0 + tx] = tile[tx][ty + i];
}

// ---------------- v [B,S,D] bf16 -> vT [B*H, HD, S] bf16 (per-head transpose) ----------------

__global__ __launch_bounds__(256) void vtrans_kernel(const unsigned short* __restrict__ in,
                                                     unsigned short* __restrict__ out) {
  __shared__ unsigned short tile[32][33];
  const int b = blockIdx.z;
  const int d0 = blockIdx.x * 32, s0 = blockIdx.y * 32;
  const int tx = threadIdx.x & 31, ty = threadIdx.x >> 5;
#pragma unroll
  for (int i = 0; i < 32; i += 8)
    tile[ty + i][tx] = in[(size_t)b * cS * cD + (size_t)(s0 + ty + i) * cD + d0 + tx];
  __syncthreads();
#pragma unroll
  for (int i = 0; i < 32; i += 8) {
    int d = d0 + ty + i;
    int h = d / cHD, hd = d % cHD;  // 32-tiles never straddle a head (HD=64)
    out[((size_t)(b * cH + h) * cHD + hd) * cS + s0 + tx] = tile[tx][ty + i];
  }
}

// ---------------- TN GEMM: C[M,N] = A[M,K] @ Bt[N,K]^T (+bias) ----------------
// Compile-time dims -> immediate-offset stores. Block tile 128x128, BK=32;
// 8 waves as 4(M) x 2(N); wave tile 32x64 -> 8 WMMA per K-step.
// Tiles are staged by the Tensor Data Mover (wave 0), not per-thread loads.

template <int MM, int NN, int KK, bool HAS_BIAS, bool OUT_F32, bool OUT_BF16>
__global__ __launch_bounds__(256) void gemm_tn_kernel(
    const unsigned short* __restrict__ A,   // MM x KK bf16
    const unsigned short* __restrict__ Bt,  // NN x KK bf16 (pre-transposed)
    const float* __restrict__ bias,         // NN (used iff HAS_BIAS)
    float* __restrict__ Cf,                 // used iff OUT_F32
    unsigned short* __restrict__ Cb) {      // used iff OUT_BF16
  __shared__ __align__(16) unsigned short lA[128 * 32];
  __shared__ __align__(16) unsigned short lBt[128 * 32];

  const int t = threadIdx.x;
  const int wave = t >> 5, lane = t & 31;
  const int waveM = wave >> 1, waveN = wave & 1;
  const int m0 = blockIdx.y * 128, n0 = blockIdx.x * 128;

  v8f acc[2][4] = {};

  for (int k0 = 0; k0 < KK; k0 += 32) {
    if (wave == 0) {  // TDM fills both LDS tiles; one instruction per tile
      tdm_load_2d(&A[(size_t)m0 * KK + k0], lds_off(lA), 128, 32, KK, KK, MM);
      tdm_load_2d(&Bt[(size_t)n0 * KK + k0], lds_off(lBt), 128, 32, KK, KK, NN);
      __builtin_amdgcn_s_wait_tensorcnt(0);
    }
    __syncthreads();

    v16bf a0 = frag_row(&lA[(waveM * 32 + 0) * 32], 32, lane);
    v16bf a1 = frag_row(&lA[(waveM * 32 + 16) * 32], 32, lane);
    v16bf b[4];
#pragma unroll
    for (int tj = 0; tj < 4; ++tj)
      b[tj] = frag_row(&lBt[(waveN * 64 + tj * 16) * 32], 32, lane);
#pragma unroll
    for (int tj = 0; tj < 4; ++tj) {
      acc[0][tj] = wmma_bf16(a0, b[tj], acc[0][tj]);
      acc[1][tj] = wmma_bf16(a1, b[tj], acc[1][tj]);
    }
    __syncthreads();
  }

  const int nn = lane & 15, mb = (lane >> 4) << 3;
  float biasv[4] = {0.f, 0.f, 0.f, 0.f};
  if (HAS_BIAS) {
#pragma unroll
    for (int tj = 0; tj < 4; ++tj)
      biasv[tj] = bias[n0 + waveN * 64 + tj * 16 + nn];
  }
#pragma unroll
  for (int ti = 0; ti < 2; ++ti)
#pragma unroll
    for (int tj = 0; tj < 4; ++tj)
#pragma unroll
      for (int r = 0; r < 8; ++r) {
        int m = m0 + waveM * 32 + ti * 16 + mb + r;
        int n = n0 + waveN * 64 + tj * 16 + nn;
        float v = acc[ti][tj][r] + biasv[tj];
        if (OUT_F32)  Cf[(size_t)m * NN + n] = v;
        if (OUT_BF16) Cb[(size_t)m * NN + n] = f2bf(v);
      }
}

// ---------------- qk logits: qk[b,h,m,n] = (q_h[m]·k_h[n]) * 0.125 + mask[m,n] ----------------
// Block tile 128(q rows) x 64(k cols), full K=64 staged once via TDM.

__global__ __launch_bounds__(256) void qk_kernel(
    const unsigned short* __restrict__ qb,  // [B,S,D] bf16
    const unsigned short* __restrict__ kb,  // [B,S,D] bf16
    const float* __restrict__ mask,         // [S,S] f32
    float* __restrict__ qkOut) {            // [B*H, S, S] f32
  const int bh = blockIdx.z;
  const int b = bh / cH, h = bh % cH;
  const int m0 = blockIdx.y * 128, n0 = blockIdx.x * 64;

  __shared__ __align__(16) unsigned short lQ[128 * 64];  // [m][kk]
  __shared__ __align__(16) unsigned short lKt[64 * 64];  // [n][kk]

  const int t = threadIdx.x;
  const int wave = t >> 5, lane = t & 31;
  const int waveM = wave >> 1, waveN = wave & 1;

  const size_t base = (size_t)b * cS * cD + (size_t)h * cHD;

  if (wave == 0) {
    tdm_load_2d(&qb[base + (size_t)m0 * cD], lds_off(lQ), 128, 64, cD, cD, cS);
    tdm_load_2d(&kb[base + (size_t)n0 * cD], lds_off(lKt), 64, 64, cD, cD, cS);
    __builtin_amdgcn_s_wait_tensorcnt(0);
  }
  __syncthreads();

  v8f acc[2][2] = {};
#pragma unroll
  for (int ks = 0; ks < 64; ks += 32) {
    v16bf a0 = frag_row(&lQ[(waveM * 32 + 0) * 64 + ks], 64, lane);
    v16bf a1 = frag_row(&lQ[(waveM * 32 + 16) * 64 + ks], 64, lane);
    v16bf b0 = frag_row(&lKt[(waveN * 32 + 0) * 64 + ks], 64, lane);
    v16bf b1 = frag_row(&lKt[(waveN * 32 + 16) * 64 + ks], 64, lane);
    acc[0][0] = wmma_bf16(a0, b0, acc[0][0]);
    acc[0][1] = wmma_bf16(a0, b1, acc[0][1]);
    acc[1][0] = wmma_bf16(a1, b0, acc[1][0]);
    acc[1][1] = wmma_bf16(a1, b1, acc[1][1]);
  }

  const int nn = lane & 15, mb = (lane >> 4) << 3;
  float* qbase = qkOut + (size_t)bh * cS * cS;
#pragma unroll
  for (int ti = 0; ti < 2; ++ti)
#pragma unroll
    for (int tj = 0; tj < 2; ++tj)
#pragma unroll
      for (int r = 0; r < 8; ++r) {
        int m = m0 + waveM * 32 + ti * 16 + mb + r;
        int n = n0 + waveN * 32 + tj * 16 + nn;
        qbase[(size_t)m * cS + n] = acc[ti][tj][r] * 0.125f + mask[(size_t)m * cS + n];
      }
}

// ---------------- row softmax statistics over qk logits ----------------

__global__ __launch_bounds__(256) void rowstats_kernel(const float* __restrict__ qk,
                                                       float* __restrict__ rowMax,
                                                       float* __restrict__ rowInv) {
  const int row = blockIdx.x;
  const float* p = qk + (size_t)row * cS;
  __shared__ float red[256];
  const int t = threadIdx.x;

  float m = -3.4e38f;
  for (int j = t; j < cS; j += 256) m = fmaxf(m, p[j]);
  red[t] = m;
  __syncthreads();
  for (int s = 128; s > 0; s >>= 1) {
    if (t < s) red[t] = fmaxf(red[t], red[t + s]);
    __syncthreads();
  }
  const float rmax = red[0];
  __syncthreads();

  float sum = 0.0f;
  for (int j = t; j < cS; j += 256) sum += expf(p[j] - rmax);
  red[t] = sum;
  __syncthreads();
  for (int s = 128; s > 0; s >>= 1) {
    if (t < s) red[t] += red[t + s];
    __syncthreads();
  }
  if (t == 0) {
    rowMax[row] = rmax;
    rowInv[row] = 1.0f / red[0];
  }
}

// ---------------- wv: on-the-fly softmax(qk) @ v_h, per (b,h) ----------------
// Block covers 128 query rows x 64 (full HD). V pre-transposed to [B*H, HD, S];
// V tiles staged by TDM, probability tiles computed into LDS by VALU.

__global__ __launch_bounds__(256) void wv_kernel(
    const float* __restrict__ qk,           // [B*H, S, S] logits
    const unsigned short* __restrict__ vT,  // [B*H, HD, S] bf16
    const float* __restrict__ rowMax,       // [B*H*S]
    const float* __restrict__ rowInv,       // [B*H*S]
    unsigned short* __restrict__ wvOut) {   // [B,S,D] bf16 (heads merged back)
  const int bh = blockIdx.z;
  const int b = bh / cH, h = bh % cH;
  const int m0 = blockIdx.y * 128;

  __shared__ __align__(16) unsigned short lP[128 * 32];  // probs bf16 [m][k]
  __shared__ __align__(16) unsigned short lV[64 * 32];   // v tile [n][k]
  __shared__ float lMax[128];
  __shared__ float lInv[128];

  const int t = threadIdx.x;
  const int wave = t >> 5, lane = t & 31;
  const int waveM = wave >> 1, waveN = wave & 1;

  if (t < 128) {
    lMax[t] = rowMax[(size_t)bh * cS + m0 + t];
    lInv[t] = rowInv[(size_t)bh * cS + m0 + t];
  }
  __syncthreads();

  const float* logits = qk + (size_t)bh * cS * cS;
  const unsigned short* vbase = vT + (size_t)bh * cHD * cS;

  v8f acc[2][2] = {};

  for (int k0 = 0; k0 < cS; k0 += 32) {
    // stage probabilities: 128x32 f32 logits -> bf16 probs (1024 float4, 4/thread)
#pragma unroll
    for (int it = 0; it < 4; ++it) {
      int idx = t + it * 256;
      int r = idx >> 3, c = (idx & 7) * 4;
      float4 lv = *(const float4*)&logits[(size_t)(m0 + r) * cS + k0 + c];
      if (k0 + 32 < cS)  // prefetch next logits tile (global_prefetch_b8)
        __builtin_prefetch(&logits[(size_t)(m0 + r) * cS + k0 + 32 + c], 0, 1);
      float mx = lMax[r], iv = lInv[r];
      unsigned p0 = f2bf(expf(lv.x - mx) * iv);
      unsigned p1 = f2bf(expf(lv.y - mx) * iv);
      unsigned p2 = f2bf(expf(lv.z - mx) * iv);
      unsigned p3 = f2bf(expf(lv.w - mx) * iv);
      unsigned* dst = (unsigned*)&lP[r * 32 + c];
      dst[0] = p0 | (p1 << 16);
      dst[1] = p2 | (p3 << 16);
    }
    if (wave == 0) {  // TDM: V tile 64(n) x 32(k)
      tdm_load_2d(&vbase[k0], lds_off(lV), 64, 32, cS, cS, cHD);
      __builtin_amdgcn_s_wait_tensorcnt(0);
    }
    __syncthreads();

    v16bf a0 = frag_row(&lP[(waveM * 32 + 0) * 32], 32, lane);
    v16bf a1 = frag_row(&lP[(waveM * 32 + 16) * 32], 32, lane);
    v16bf b0 = frag_row(&lV[(waveN * 32 + 0) * 32], 32, lane);
    v16bf b1 = frag_row(&lV[(waveN * 32 + 16) * 32], 32, lane);
    acc[0][0] = wmma_bf16(a0, b0, acc[0][0]);
    acc[0][1] = wmma_bf16(a0, b1, acc[0][1]);
    acc[1][0] = wmma_bf16(a1, b0, acc[1][0]);
    acc[1][1] = wmma_bf16(a1, b1, acc[1][1]);
    __syncthreads();
  }

  const int nn = lane & 15, mb = (lane >> 4) << 3;
#pragma unroll
  for (int ti = 0; ti < 2; ++ti)
#pragma unroll
    for (int tj = 0; tj < 2; ++tj)
#pragma unroll
      for (int r = 0; r < 8; ++r) {
        int m = m0 + waveM * 32 + ti * 16 + mb + r;
        int n = waveN * 32 + tj * 16 + nn;  // 0..63 within head
        wvOut[(size_t)b * cS * cD + (size_t)m * cD + h * cHD + n] = f2bf(acc[ti][tj][r]);
      }
}

// ---------------- host-side launch ----------------

extern "C" void kernel_launch(void* const* d_in, const int* in_sizes, int n_in,
                              void* d_out, int out_size, void* d_ws, size_t ws_size,
                              hipStream_t stream) {
  (void)in_sizes; (void)n_in; (void)out_size; (void)ws_size;

  const float* x    = (const float*)d_in[0];
  const float* mask = (const float*)d_in[1];
  const float* Wq   = (const float*)d_in[2];
  const float* bq   = (const float*)d_in[3];
  const float* Wk   = (const float*)d_in[4];
  const float* Wv   = (const float*)d_in[5];
  const float* bv   = (const float*)d_in[6];
  const float* Wo   = (const float*)d_in[7];
  const float* bo   = (const float*)d_in[8];

  const int M  = cB * cS;   // 8192
  const int nX = M * cD;    // 8,388,608
  const int nRows = cB * cH * cS;

  // d_out: [out (B*S*D f32), qk (B*H*S*S f32)]
  float* outPtr = (float*)d_out;
  float* qkPtr  = (float*)d_out + (size_t)nX;

  // workspace carve-out (bf16 stored as ushort)
  char* w = (char*)d_ws;
  auto take = [&](size_t bytes) {
    char* p = w;
    w += (bytes + 255) & ~(size_t)255;
    return p;
  };
  unsigned short* xb  = (unsigned short*)take((size_t)nX * 2);
  unsigned short* wqt = (unsigned short*)take((size_t)cD * cD * 2);
  unsigned short* wkt = (unsigned short*)take((size_t)cD * cD * 2);
  unsigned short* wvt = (unsigned short*)take((size_t)cD * cD * 2);
  unsigned short* wot = (unsigned short*)take((size_t)cD * cD * 2);
  unsigned short* qb  = (unsigned short*)take((size_t)nX * 2);
  unsigned short* kbb = (unsigned short*)take((size_t)nX * 2);
  unsigned short* vbb = (unsigned short*)take((size_t)nX * 2);
  unsigned short* vTT = (unsigned short*)take((size_t)nX * 2);
  unsigned short* wvo = (unsigned short*)take((size_t)nX * 2);
  float* rowMax = (float*)take((size_t)nRows * 4);
  float* rowInv = (float*)take((size_t)nRows * 4);

  // 1) x -> bf16 ; weights -> bf16 transposed [N,K]
  cvt_f32_bf16_kernel<<<(nX + 255) / 256, 256, 0, stream>>>(x, xb, nX);
  dim3 gT(cD / 32, cD / 32, 1);
  cvt_t_kernel<<<gT, 256, 0, stream>>>(Wq, wqt, cD, cD);
  cvt_t_kernel<<<gT, 256, 0, stream>>>(Wk, wkt, cD, cD);
  cvt_t_kernel<<<gT, 256, 0, stream>>>(Wv, wvt, cD, cD);
  cvt_t_kernel<<<gT, 256, 0, stream>>>(Wo, wot, cD, cD);

  // 2) q/k/v projections (bf16 outputs)
  dim3 gGemm(cD / 128, M / 128, 1);
  gemm_tn_kernel<8192, 1024, 1024, true,  false, true><<<gGemm, 256, 0, stream>>>(xb, wqt, bq, nullptr, qb);
  gemm_tn_kernel<8192, 1024, 1024, false, false, true><<<gGemm, 256, 0, stream>>>(xb, wkt, nullptr, nullptr, kbb);
  gemm_tn_kernel<8192, 1024, 1024, true,  false, true><<<gGemm, 256, 0, stream>>>(xb, wvt, bv, nullptr, vbb);

  // 3) per-head transpose of v
  dim3 gVT(cD / 32, cS / 32, cB);
  vtrans_kernel<<<gVT, 256, 0, stream>>>(vbb, vTT);

  // 4) qk logits (+mask) into d_out
  dim3 gQK(cS / 64, cS / 128, cB * cH);
  qk_kernel<<<gQK, 256, 0, stream>>>(qb, kbb, mask, qkPtr);

  // 5) softmax row statistics
  rowstats_kernel<<<nRows, 256, 0, stream>>>(qkPtr, rowMax, rowInv);

  // 6) wv = softmax(qk) @ v  (on-the-fly probabilities)
  dim3 gWV(1, cS / 128, cB * cH);
  wv_kernel<<<gWV, 256, 0, stream>>>(qkPtr, vTT, rowMax, rowInv, wvo);

  // 7) output projection into d_out
  gemm_tn_kernel<8192, 1024, 1024, true, true, false><<<gGemm, 256, 0, stream>>>(wvo, wot, bo, outPtr, nullptr);
}
